// SparseSoftHyperedgeGeneration_21869973471344
// MI455X (gfx1250) — matrix operand order repaired and checked by reading, not compiled, not run on hardware
//
#include <hip/hip_runtime.h>
#include <hip/hip_bf16.h>

typedef _Float16 f16;
typedef _Float16 v16h __attribute__((ext_vector_type(16)));
typedef _Float16 v8h  __attribute__((ext_vector_type(8)));
typedef float    v8f  __attribute__((ext_vector_type(8)));

#define BDIM 8
#define NDIM 4096
#define DDIM 256
#define EDIM 256
#define EFIX 16
#define KTOP 64

// ---------------- K1: ctx = [mean_n X, max_n X] -> [B, 2D] ----------------
__global__ void ctx_kernel(const float* __restrict__ X, float* __restrict__ ctx) {
    const int b = blockIdx.x;
    const int d = threadIdx.x;
    const float* xp = X + (size_t)b * NDIM * DDIM + d;
    float s = 0.0f, m = -__builtin_inff();
    for (int n = 0; n < NDIM; ++n) {
        float v = xp[(size_t)n * DDIM];
        s += v;
        m = fmaxf(m, v);
    }
    ctx[b * 2 * DDIM + d]        = s * (1.0f / NDIM);
    ctx[b * 2 * DDIM + DDIM + d] = m;
}

// ------------- K1b: WpT[c,k] = (f16)Wp[k,c]  (one-time 128 KB prep) --------
__global__ void wpt_kernel(const float* __restrict__ Wp, f16* __restrict__ WpT) {
    const int k = blockIdx.x;      // 256
    const int c = threadIdx.x;     // 256
    WpT[(size_t)c * DDIM + k] = (f16)Wp[(size_t)k * DDIM + c];
}

// ---- K2: P[b, j] = base[j] + bc[j] + sum_k ctx[b,k]*Wc[k,j]  (f16 out) ----
// Wc (128 MB) is the dominant HBM stream; each element read exactly once.
// Stride between k-iterations is 256 KB -> software prefetch helps.
__global__ void proto_kernel(const float* __restrict__ ctx,
                             const float* __restrict__ base,
                             const float* __restrict__ Wc,
                             const float* __restrict__ bc,
                             f16* __restrict__ P) {
    __shared__ float sctx[BDIM * 2 * DDIM];   // 16 KB of the 320 KB LDS
    const int tid = threadIdx.x;
    for (int i = tid; i < BDIM * 2 * DDIM; i += 256) sctx[i] = ctx[i];
    __syncthreads();

    const int j = blockIdx.x * 256 + tid;     // 0..65535
    float acc[BDIM];
#pragma unroll
    for (int bb = 0; bb < BDIM; ++bb) acc[bb] = 0.0f;

    for (int k = 0; k < 2 * DDIM; ++k) {
        if (k + 8 < 2 * DDIM)   // global_prefetch_b8 of the next strided line
            __builtin_prefetch(&Wc[(size_t)(k + 8) * (EDIM * DDIM) + j], 0, 1);
        float w = Wc[(size_t)k * (EDIM * DDIM) + j];
#pragma unroll
        for (int bb = 0; bb < BDIM; ++bb)
            acc[bb] = fmaf(sctx[bb * 2 * DDIM + k], w, acc[bb]);
    }
    const float c = base[j] + bc[j];
#pragma unroll
    for (int bb = 0; bb < BDIM; ++bb)
        P[(size_t)bb * EDIM * DDIM + j] = (f16)(acc[bb] + c);
}

// ---- fragment loaders (ISA 7.12.2 layouts: A 16x32 f16, B 32x16 f16) ------
__device__ __forceinline__ v16h load_a_f32cvt(const float* __restrict__ row, int k0, int khA) {
    v16h a;
    const float* p0 = row + k0 + khA;
    const float* p1 = p0 + 16;
#pragma unroll
    for (int i = 0; i < 8; ++i) { a[i] = (f16)p0[i]; a[8 + i] = (f16)p1[i]; }
    return a;
}
__device__ __forceinline__ v16h load_a_f16(const f16* __restrict__ row, int k0, int khA) {
    v8h lo = *(const v8h*)(row + k0 + khA);
    v8h hi = *(const v8h*)(row + k0 + 16 + khA);
    v16h a;
#pragma unroll
    for (int i = 0; i < 8; ++i) { a[i] = lo[i]; a[8 + i] = hi[i]; }
    return a;
}
__device__ __forceinline__ v16h load_b_f16(const f16* __restrict__ row, int k0, int khB) {
    v8h lo = *(const v8h*)(row + k0 + khB);
    v8h hi = *(const v8h*)(row + k0 + 16 + khB);
    v16h b;
#pragma unroll
    for (int i = 0; i < 8; ++i) { b[i] = lo[i]; b[8 + i] = hi[i]; }
    return b;
}

// ------------- K3: X_proj = X @ Wp + bp  (WMMA f16, f32 accum) -------------
// Each wave computes a 16x64 strip. Per k-step: load A + all 4 B fragments
// first (one VMEM clause), then issue the 4 WMMAs -> compiler emits
// descending partial s_wait_loadcnt, overlapping matrix pipe with VMEM,
// without the register pressure of cross-iteration double buffering.
__global__ void xproj_kernel(const float* __restrict__ X,
                             const f16*   __restrict__ WpT,
                             const float* __restrict__ bp,
                             f16* __restrict__ XP) {
    const int lane   = threadIdx.x & 31;
    const int wave   = threadIdx.x >> 5;
    const int strip  = blockIdx.x * 4 + wave;   // 8192 strips
    const int m0     = (strip >> 2) * 16;       // row tile (0..2047)
    const int c0     = (strip & 3) * 64;        // 64-col group
    const int laneHi = lane >> 4;
    const int lanelo = lane & 15;
    const int khA    = laneHi ? 8 : 0;          // A K-half offset
    const int khB    = laneHi ? 8 : 0;          // B K-half offset

    const float* Arow = X + (size_t)(m0 + lanelo) * DDIM;
    const f16* Brow[4];
#pragma unroll
    for (int t = 0; t < 4; ++t)
        Brow[t] = WpT + (size_t)(c0 + t * 16 + lanelo) * DDIM;

    v8f cacc[4] = {};
    for (int k0 = 0; k0 < DDIM; k0 += 32) {
        v16h a = load_a_f32cvt(Arow, k0, khA);
        v16h bfr[4];
#pragma unroll
        for (int t = 0; t < 4; ++t) bfr[t] = load_b_f16(Brow[t], k0, khB);
#pragma unroll
        for (int t = 0; t < 4; ++t)
            cacc[t] = __builtin_amdgcn_wmma_f32_16x16x32_f16(
                false, a, false, bfr[t], (short)0, cacc[t], false, false);
    }

    const int row0 = m0 + (laneHi ? 8 : 0);
#pragma unroll
    for (int t = 0; t < 4; ++t) {
        const int col = c0 + t * 16 + lanelo;
        const float bpv = bp[col];
#pragma unroll
        for (int r = 0; r < 8; ++r)
            XP[(size_t)(row0 + r) * DDIM + col] = (f16)(cacc[t][r] + bpv);
    }
}

// ------- K4: logits[b] = X_proj[b] @ P[b]^T / 32  (WMMA f16 -> d_out) ------
// Same load-all-then-compute-all k-step; B fragment column e0+n is a
// contiguous row of P -> pure b128 loads.
__global__ void logits_kernel(const f16* __restrict__ XP,
                              const f16* __restrict__ P,
                              float* __restrict__ L) {
    const int lane   = threadIdx.x & 31;
    const int wave   = threadIdx.x >> 5;
    const int strip  = blockIdx.x * 4 + wave;   // 8192 strips total
    const int b      = strip >> 10;             // 1024 strips per batch
    const int rem    = strip & 1023;
    const int m0     = (rem >> 2) * 16;         // row tile (0..255)
    const int e0     = (rem & 3) * 64;          // 64-col group
    const int laneHi = lane >> 4;
    const int lanelo = lane & 15;
    const int khA    = laneHi ? 8 : 0;
    const int khB    = laneHi ? 8 : 0;

    const f16* Xrow = XP + (size_t)b * NDIM * DDIM + (size_t)(m0 + lanelo) * DDIM;
    const f16* Pb   = P  + (size_t)b * EDIM * DDIM;
    const f16* Prow[4];
#pragma unroll
    for (int t = 0; t < 4; ++t)
        Prow[t] = Pb + (size_t)(e0 + t * 16 + lanelo) * DDIM;

    v8f cacc[4] = {};
    for (int k0 = 0; k0 < DDIM; k0 += 32) {
        v16h a = load_a_f16(Xrow, k0, khA);
        v16h bfr[4];
#pragma unroll
        for (int t = 0; t < 4; ++t) bfr[t] = load_b_f16(Prow[t], k0, khB);
#pragma unroll
        for (int t = 0; t < 4; ++t)
            cacc[t] = __builtin_amdgcn_wmma_f32_16x16x32_f16(
                false, a, false, bfr[t], (short)0, cacc[t], false, false);
    }

    float* Lb = L + (size_t)b * NDIM * EDIM;
    const int row0 = m0 + (laneHi ? 8 : 0);
#pragma unroll
    for (int t = 0; t < 4; ++t) {
        const int col = e0 + t * 16 + lanelo;
#pragma unroll
        for (int r = 0; r < 8; ++r)
            Lb[(size_t)(row0 + r) * EDIM + col] = cacc[t][r] * 0.03125f; // /(sqrt(64)*4)
    }
}

// ----------------- K5: scores[b,e] = sum_n logits[b,n,e] ------------------
__global__ void scores_kernel(const float* __restrict__ L, float* __restrict__ scores) {
    const int b = blockIdx.x, e = threadIdx.x;
    const float* Lb = L + (size_t)b * NDIM * EDIM + e;
    float s = 0.0f;
    for (int n = 0; n < NDIM; ++n) s += Lb[(size_t)n * EDIM];
    scores[b * EDIM + e] = s;
}

// --------- K6: mask = first 16 ones + top-64 of scores[16:] ---------------
// rank-by-count reproduces lax.top_k's lowest-index tie-break deterministically.
__global__ void mask_kernel(const float* __restrict__ scores, float* __restrict__ mask) {
    __shared__ float sc[EDIM];
    const int b = blockIdx.x, e = threadIdx.x;
    sc[e] = scores[b * EDIM + e];
    __syncthreads();
    float mv;
    if (e < EFIX) {
        mv = 1.0f;
    } else {
        const float se = sc[e];
        int rank = 0;
        for (int j = EFIX; j < EDIM; ++j) {
            float sj = sc[j];
            if (sj > se || (sj == se && j < e)) ++rank;
        }
        mv = (rank < KTOP) ? 1.0f : 0.0f;
    }
    mask[b * EDIM + e] = mv;
}

// --------- K7: in-place softmax over nodes of (logits * mask) -------------
// Online max/sum pass, then normalize; masked columns become uniform 1/N
// exactly like the reference (softmax of an all-zero column).
__global__ void softmax_kernel(float* __restrict__ L, const float* __restrict__ mask) {
    const int b = blockIdx.x, e = threadIdx.x;
    const float mv = mask[b * EDIM + e];
    float* Lb = L + (size_t)b * NDIM * EDIM + e;
    float mx = -__builtin_inff(), s = 0.0f;
    for (int n = 0; n < NDIM; ++n) {
        float v = Lb[(size_t)n * EDIM] * mv;
        if (v > mx) { s = s * __expf(mx - v) + 1.0f; mx = v; }
        else        { s += __expf(v - mx); }
    }
    const float inv = 1.0f / s;
    for (int n = 0; n < NDIM; ++n) {
        float v = Lb[(size_t)n * EDIM] * mv;
        Lb[(size_t)n * EDIM] = __expf(v - mx) * inv;
    }
}

extern "C" void kernel_launch(void* const* d_in, const int* in_sizes, int n_in,
                              void* d_out, int out_size, void* d_ws, size_t ws_size,
                              hipStream_t stream) {
    (void)in_sizes; (void)n_in; (void)out_size; (void)ws_size;
    const float* X    = (const float*)d_in[0];
    const float* base = (const float*)d_in[1];
    const float* Wc   = (const float*)d_in[2];
    const float* bc   = (const float*)d_in[3];
    const float* Wp   = (const float*)d_in[4];
    const float* bp   = (const float*)d_in[5];
    float* out = (float*)d_out;

    char* ws = (char*)d_ws;
    float* ctx    = (float*)(ws);                           // 16 KB
    float* scores = (float*)(ws + 16384);                   // 8 KB
    float* mask   = (float*)(ws + 24576);                   // 8 KB
    f16*   P      = (f16*)(ws + 32768);                     // 1 MB   [B,E,D] f16
    f16*   XP     = (f16*)(ws + 32768 + 1048576);           // 16 MB  [B,N,D] f16
    f16*   WpT    = (f16*)(ws + 32768 + 1048576 + 16777216);// 128 KB [D,D]   f16 (transposed)

    ctx_kernel    <<<BDIM, 256, 0, stream>>>(X, ctx);
    wpt_kernel    <<<DDIM, DDIM, 0, stream>>>(Wp, WpT);
    proto_kernel  <<<(EDIM * DDIM) / 256, 256, 0, stream>>>(ctx, base, Wc, bc, P);
    xproj_kernel  <<<(BDIM * NDIM / 16) * (DDIM / 64) / 4, 128, 0, stream>>>(X, WpT, bp, XP);
    logits_kernel <<<(BDIM * (NDIM / 16) * (EDIM / 64)) / 4, 128, 0, stream>>>(XP, P, out);
    scores_kernel <<<BDIM, EDIM, 0, stream>>>(out, scores);
    mask_kernel   <<<BDIM, EDIM, 0, stream>>>(scores, mask);
    softmax_kernel<<<BDIM, EDIM, 0, stream>>>(out, mask);
}
